// LaDCastSingleTransformerBlock_3410204033546
// MI455X (gfx1250) — matrix operand order repaired and checked by reading, not compiled, not run on hardware
//
#include <hip/hip_runtime.h>
#include <hip/hip_bf16.h>

#define DEV __device__ __forceinline__

typedef __attribute__((ext_vector_type(16))) __bf16 v16bf;
typedef __attribute__((ext_vector_type(8)))  __bf16 v8bf;
typedef __attribute__((ext_vector_type(8)))  float  v8f;

// ---- problem dims ----
constexpr int Bn = 2, S_IMG = 1536, S_TXT = 512, Sq = 2048, Hh = 16;
constexpr int HID = 1024, MLPD = 4096, TOK = Bn * Sq;   // 4096 tokens

// ---- workspace layout (bytes) ----
constexpr size_t OFF_EMB   = 0;                                   // [2,3072] f32
constexpr size_t OFF_NX    = 32768;                               // [4096,1024] bf16
constexpr size_t OFF_WQKV  = OFF_NX   + (size_t)TOK*HID*2;        // [3072][1024] bf16
constexpr size_t OFF_WMLP  = OFF_WQKV + (size_t)3*HID*HID*2;      // [4096][1024] bf16
constexpr size_t OFF_WOUT  = OFF_WMLP + (size_t)MLPD*HID*2;       // [1024][5120] bf16
constexpr size_t OFF_QKV   = OFF_WOUT + (size_t)HID*(HID+MLPD)*2; // [4096,3072] f32
constexpr size_t OFF_Q     = OFF_QKV  + (size_t)TOK*3*HID*4;      // [B,H,S,64] bf16
constexpr size_t OFF_K     = OFF_Q    + (size_t)TOK*HID*2;
constexpr size_t OFF_VT    = OFF_K    + (size_t)TOK*HID*2;        // [B,H,64,S] bf16
constexpr size_t OFF_O     = OFF_VT   + (size_t)TOK*HID*2;        // [4096,1024] bf16
constexpr size_t OFF_MLPA  = OFF_O    + (size_t)TOK*HID*2;        // [4096,4096] bf16

DEV __bf16 f2bf(float f) {
  union { float f; unsigned u; } a; a.f = f;
  unsigned u = (a.u + 0x7FFFu + ((a.u >> 16) & 1u)) >> 16;
  union { unsigned short s; __bf16 b; } o; o.s = (unsigned short)u;
  return o.b;
}

// A fragment (16x32 bf16): lane m = lane&15; low lanes hold K 0..15, high lanes 16..31
DEV v16bf load_a16(const __bf16* rowp, int kbase, int lhalf) {
  v8bf lo = *(const v8bf*)(rowp + kbase + lhalf * 16);
  v8bf hi = *(const v8bf*)(rowp + kbase + lhalf * 16 + 8);
  return __builtin_shufflevector(lo, hi, 0,1,2,3,4,5,6,7,8,9,10,11,12,13,14,15);
}
// B fragment (32x16 bf16): lane n = lane&15 holds 16 contiguous K (low lanes K0..15, high K16..31)
DEV v16bf load_b16(const __bf16* colp, int kbase, int lhalf) {
  return *(const v16bf*)(colp + kbase + lhalf * 16);
}
DEV v8f wmma_bf16(v16bf a, v16bf b, v8f c) {
  return __builtin_amdgcn_wmma_f32_16x16x32_bf16(false, a, false, b, (short)0, c, false, false);
}

// ---------- weight fp32 [K,N] -> bf16 [N,K] ----------
__global__ void k_transpose_bf16(const float* __restrict__ in, __bf16* __restrict__ out,
                                 int K, int N) {
  int idx = blockIdx.x * blockDim.x + threadIdx.x;
  int n = idx / K, k = idx % K;
  out[idx] = f2bf(in[(size_t)k * N + n]);
}

// ---------- AdaLN params: emb = silu(temb) @ w_norm + b_norm ----------
__global__ void k_ada(const float* __restrict__ temb, const float* __restrict__ wn,
                      const float* __restrict__ bnv, float* __restrict__ emb) {
  int idx = blockIdx.x * blockDim.x + threadIdx.x;     // [0, 2*3072)
  int b = idx / 3072, c = idx % 3072;
  const float* t = temb + b * HID;
  float s = 0.f;
  for (int k = 0; k < HID; ++k) {
    float x = t[k];
    s += (x / (1.f + __expf(-x))) * wn[(size_t)k * 3072 + c];
  }
  emb[idx] = s + bnv[c];
}

// ---------- LayerNorm + modulate -> nx bf16 ----------
__global__ void k_ln(const float* __restrict__ hid, const float* __restrict__ enc,
                     const float* __restrict__ emb, __bf16* __restrict__ nx) {
  int tok = blockIdx.x;
  int b = tok / Sq, s = tok % Sq;
  const float* row = (s < S_IMG) ? hid + (size_t)(b * S_IMG + s) * HID
                                 : enc + (size_t)(b * S_TXT + (s - S_IMG)) * HID;
  int tid = threadIdx.x;
  float v[4], ps = 0.f, ps2 = 0.f;
  for (int i = 0; i < 4; ++i) { float x = row[tid + 256 * i]; v[i] = x; ps += x; ps2 += x * x; }
  __shared__ float r1[256], r2[256];
  r1[tid] = ps; r2[tid] = ps2; __syncthreads();
  for (int o = 128; o > 0; o >>= 1) {
    if (tid < o) { r1[tid] += r1[tid + o]; r2[tid] += r2[tid + o]; }
    __syncthreads();
  }
  float mu = r1[0] * (1.f / HID);
  float rinv = rsqrtf(r2[0] * (1.f / HID) - mu * mu + 1e-6f);
  __bf16* orow = nx + (size_t)tok * HID;
  const float* eb = emb + b * 3072;
  for (int i = 0; i < 4; ++i) {
    int c = tid + 256 * i;
    orow[c] = f2bf((v[i] - mu) * rinv * (1.f + eb[HID + c]) + eb[c]);
  }
}

// ---------- QKV GEMM: 32x64 wave tile; nx[4096,1024] @ wqkv_t -> qkv f32 [4096,3072] ----------
__global__ void k_gemm_qkv(const __bf16* __restrict__ nx, const __bf16* __restrict__ wt,
                           const float* __restrict__ bq, const float* __restrict__ bk,
                           const float* __restrict__ bv, float* __restrict__ qkv) {
  int wav = (blockIdx.x * blockDim.x + threadIdx.x) >> 5;
  int lane = threadIdx.x & 31, lhalf = lane >> 4, lm = lane & 15;
  const int MT = TOK / 32;                    // 128
  int m0 = (wav % MT) * 32, n0 = (wav / MT) * 64;
  const __bf16* arow0 = nx + (size_t)(m0 + lm) * HID;
  const __bf16* arow1 = nx + (size_t)(m0 + 16 + lm) * HID;
  // warm-up prefetch of the streamed rows (outside the hot loop: no register pressure)
  __builtin_prefetch(arow0, 0, 3);
  __builtin_prefetch(arow1, 0, 3);
  v8f acc0[4], acc1[4];
  for (int t = 0; t < 4; ++t) for (int r = 0; r < 8; ++r) { acc0[t][r] = 0.f; acc1[t][r] = 0.f; }
  for (int kk = 0; kk < HID; kk += 32) {
    v16bf a0 = load_a16(arow0, kk, lhalf);
    v16bf a1 = load_a16(arow1, kk, lhalf);
    #pragma unroll
    for (int t = 0; t < 4; ++t) {
      const __bf16* bp = wt + (size_t)(n0 + t * 16 + lm) * HID;
      v16bf bfr = load_b16(bp, kk, lhalf);
      acc0[t] = wmma_bf16(a0, bfr, acc0[t]);
      acc1[t] = wmma_bf16(a1, bfr, acc1[t]);
    }
  }
  for (int t = 0; t < 4; ++t) {
    int col = n0 + t * 16 + lm;
    float bias = (col < 1024) ? bq[col] : (col < 2048) ? bk[col - 1024] : bv[col - 2048];
    for (int r = 0; r < 8; ++r) {
      qkv[(size_t)(m0 + r + 8 * lhalf) * 3072 + col]      = acc0[t][r] + bias;
      qkv[(size_t)(m0 + 16 + r + 8 * lhalf) * 3072 + col] = acc1[t][r] + bias;
    }
  }
}

// ---------- MLP GEMM + gelu(tanh) -> bf16 [4096,4096] ----------
__global__ void k_gemm_mlp(const __bf16* __restrict__ nx, const __bf16* __restrict__ wt,
                           const float* __restrict__ bm, __bf16* __restrict__ mlp) {
  int wav = (blockIdx.x * blockDim.x + threadIdx.x) >> 5;
  int lane = threadIdx.x & 31, lhalf = lane >> 4, lm = lane & 15;
  const int MT = TOK / 32;
  int m0 = (wav % MT) * 32, n0 = (wav / MT) * 64;
  const __bf16* arow0 = nx + (size_t)(m0 + lm) * HID;
  const __bf16* arow1 = nx + (size_t)(m0 + 16 + lm) * HID;
  __builtin_prefetch(arow0, 0, 3);
  __builtin_prefetch(arow1, 0, 3);
  v8f acc0[4], acc1[4];
  for (int t = 0; t < 4; ++t) for (int r = 0; r < 8; ++r) { acc0[t][r] = 0.f; acc1[t][r] = 0.f; }
  for (int kk = 0; kk < HID; kk += 32) {
    v16bf a0 = load_a16(arow0, kk, lhalf);
    v16bf a1 = load_a16(arow1, kk, lhalf);
    #pragma unroll
    for (int t = 0; t < 4; ++t) {
      const __bf16* bp = wt + (size_t)(n0 + t * 16 + lm) * HID;
      v16bf bfr = load_b16(bp, kk, lhalf);
      acc0[t] = wmma_bf16(a0, bfr, acc0[t]);
      acc1[t] = wmma_bf16(a1, bfr, acc1[t]);
    }
  }
  for (int t = 0; t < 4; ++t) {
    int col = n0 + t * 16 + lm;
    float bias = bm[col];
    for (int r = 0; r < 8; ++r) {
      float x0 = acc0[t][r] + bias;
      float x1 = acc1[t][r] + bias;
      float g0 = 0.5f * x0 * (1.f + tanhf(0.7978845608f * (x0 + 0.044715f * x0 * x0 * x0)));
      float g1 = 0.5f * x1 * (1.f + tanhf(0.7978845608f * (x1 + 0.044715f * x1 * x1 * x1)));
      mlp[(size_t)(m0 + r + 8 * lhalf) * MLPD + col]      = f2bf(g0);
      mlp[(size_t)(m0 + 16 + r + 8 * lhalf) * MLPD + col] = f2bf(g1);
    }
  }
}

// ---------- RMS-norm + RoPE; Q,K -> [B,H,S,64] bf16 ; V -> [B,H,64,S] bf16 ----------
__global__ void k_rmsrope(const float* __restrict__ qkv, const float* __restrict__ gq,
                          const float* __restrict__ gk,
                          const float* __restrict__ rc, const float* __restrict__ rsn,
                          const float* __restrict__ cc, const float* __restrict__ csn,
                          __bf16* __restrict__ q, __bf16* __restrict__ k,
                          __bf16* __restrict__ vt) {
  int unit = (blockIdx.x * blockDim.x + threadIdx.x) >> 5;   // b*S*H units
  int lane = threadIdx.x & 31;
  int b = unit / (Sq * Hh);
  int rem = unit % (Sq * Hh);
  int s = rem / Hh, h = rem % Hh;
  const float* base = qkv + (size_t)(b * Sq + s) * 3072 + h * 64;
  int d0 = lane * 2;
  float q0 = base[d0], q1 = base[d0 + 1];
  float k0 = base[1024 + d0], k1 = base[1024 + d0 + 1];
  float v0 = base[2048 + d0], v1 = base[2048 + d0 + 1];
  float sq = q0 * q0 + q1 * q1, sk = k0 * k0 + k1 * k1;
  for (int o = 16; o; o >>= 1) { sq += __shfl_xor(sq, o, 32); sk += __shfl_xor(sk, o, 32); }
  float rq = rsqrtf(sq * (1.f / 64.f) + 1e-7f), rk = rsqrtf(sk * (1.f / 64.f) + 1e-7f);
  q0 *= rq * gq[d0]; q1 *= rq * gq[d0 + 1];
  k0 *= rk * gk[d0]; k1 *= rk * gk[d0 + 1];
  const float *Cp, *Sp; int so;
  if (s < S_IMG) { Cp = rc; Sp = rsn; so = s; } else { Cp = cc; Sp = csn; so = s - S_IMG; }
  float c0 = Cp[so * 64 + d0], c1 = Cp[so * 64 + d0 + 1];
  float s0 = Sp[so * 64 + d0], s1 = Sp[so * 64 + d0 + 1];
  size_t rowq = ((size_t)(b * Hh + h) * Sq + s) * 64;
  q[rowq + d0]     = f2bf(q0 * c0 - q1 * s0);
  q[rowq + d0 + 1] = f2bf(q1 * c1 + q0 * s1);
  k[rowq + d0]     = f2bf(k0 * c0 - k1 * s0);
  k[rowq + d0 + 1] = f2bf(k1 * c1 + k0 * s1);
  size_t vb = (size_t)(b * Hh + h) * 64;
  vt[(vb + d0) * Sq + s]     = f2bf(v0);
  vt[(vb + d0 + 1) * Sq + s] = f2bf(v1);
}

// ---------- Flash attention: 1 wave per (b,h, 16-query tile) ----------
__global__ void k_attn(const __bf16* __restrict__ q, const __bf16* __restrict__ k,
                       const __bf16* __restrict__ vt, __bf16* __restrict__ o) {
  __shared__ __bf16 plds[8][16 * 32];
  int wav = threadIdx.x >> 5, lane = threadIdx.x & 31;
  int lhalf = lane >> 4, lm = lane & 15;
  int gt = blockIdx.x * 8 + wav;
  int qt = gt % (Sq / 16);
  int bh = gt / (Sq / 16);
  const __bf16* qrow = q + ((size_t)bh * Sq + qt * 16 + lm) * 64;
  v16bf aQ0 = load_a16(qrow, 0, lhalf);
  v16bf aQ1 = load_a16(qrow, 32, lhalf);
  const __bf16* kbase = k + (size_t)bh * Sq * 64;
  const __bf16* vbase = vt + (size_t)bh * 64 * Sq;
  float mR[8], lR[8];
  v8f acc[4];
  for (int r = 0; r < 8; ++r) { mR[r] = -1e30f; lR[r] = 0.f; }
  for (int t = 0; t < 4; ++t) for (int r = 0; r < 8; ++r) acc[t][r] = 0.f;
  __bf16* lds = plds[wav];
  for (int kb = 0; kb < Sq; kb += 32) {
    v8f s0, s1;
    for (int r = 0; r < 8; ++r) { s0[r] = 0.f; s1[r] = 0.f; }
    const __bf16* kr0 = kbase + (size_t)(kb + lm) * 64;
    const __bf16* kr1 = kbase + (size_t)(kb + 16 + lm) * 64;
    s0 = wmma_bf16(aQ0, load_b16(kr0, 0, lhalf), s0);
    s0 = wmma_bf16(aQ1, load_b16(kr0, 32, lhalf), s0);
    s1 = wmma_bf16(aQ0, load_b16(kr1, 0, lhalf), s1);
    s1 = wmma_bf16(aQ1, load_b16(kr1, 32, lhalf), s1);
    #pragma unroll
    for (int r = 0; r < 8; ++r) {
      float a0 = s0[r] * 0.125f, a1 = s1[r] * 0.125f;
      float cm = fmaxf(a0, a1);
      for (int ox = 8; ox; ox >>= 1) cm = fmaxf(cm, __shfl_xor(cm, ox, 32));
      float mn = fmaxf(mR[r], cm);
      float alpha = __expf(mR[r] - mn);
      float p0 = __expf(a0 - mn), p1 = __expf(a1 - mn);
      float rs = p0 + p1;
      for (int ox = 8; ox; ox >>= 1) rs += __shfl_xor(rs, ox, 32);
      lR[r] = lR[r] * alpha + rs;
      mR[r] = mn;
      acc[0][r] *= alpha; acc[1][r] *= alpha; acc[2][r] *= alpha; acc[3][r] *= alpha;
      int row = r + 8 * lhalf;                 // C-layout row for this lane
      lds[row * 32 + lm]      = f2bf(p0);
      lds[row * 32 + 16 + lm] = f2bf(p1);
    }
    // transpose probs to A-layout through LDS (same-wave DS ops are in order)
    v8bf lo = *(const v8bf*)(lds + lm * 32 + lhalf * 16);
    v8bf hi = *(const v8bf*)(lds + lm * 32 + lhalf * 16 + 8);
    v16bf pA = __builtin_shufflevector(lo, hi, 0,1,2,3,4,5,6,7,8,9,10,11,12,13,14,15);
    #pragma unroll
    for (int t = 0; t < 4; ++t) {
      const __bf16* vrow = vbase + (size_t)(t * 16 + lm) * Sq;
      v16bf bv = *(const v16bf*)(vrow + kb + lhalf * 16);
      acc[t] = wmma_bf16(pA, bv, acc[t]);
    }
  }
  int b = bh / Hh, h = bh % Hh;
  for (int r = 0; r < 8; ++r) {
    float inv = 1.f / lR[r];
    int m = qt * 16 + r + 8 * lhalf;
    __bf16* orow = o + ((size_t)(b * Sq + m)) * HID + h * 64;
    for (int t = 0; t < 4; ++t)
      orow[t * 16 + lm] = f2bf(acc[t][r] * inv);
  }
}

// ---------- out GEMM: 32x64 tile; concat(o,mlp)[4096,5120] @ wout_t + gate/residual/split ----------
__global__ void k_gemm_out(const __bf16* __restrict__ ob, const __bf16* __restrict__ mlp,
                           const __bf16* __restrict__ wt, const float* __restrict__ bo,
                           const float* __restrict__ emb, const float* __restrict__ hid,
                           const float* __restrict__ enc, float* __restrict__ out) {
  int wav = (blockIdx.x * blockDim.x + threadIdx.x) >> 5;
  int lane = threadIdx.x & 31, lhalf = lane >> 4, lm = lane & 15;
  const int MT = TOK / 32;
  int m0 = (wav % MT) * 32, n0 = (wav / MT) * 64;
  const __bf16* oR0 = ob  + (size_t)(m0 + lm) * HID;
  const __bf16* oR1 = ob  + (size_t)(m0 + 16 + lm) * HID;
  const __bf16* mR0 = mlp + (size_t)(m0 + lm) * MLPD;
  const __bf16* mR1 = mlp + (size_t)(m0 + 16 + lm) * MLPD;
  __builtin_prefetch(oR0, 0, 3);
  __builtin_prefetch(oR1, 0, 3);
  v8f acc0[4], acc1[4];
  for (int t = 0; t < 4; ++t) for (int r = 0; r < 8; ++r) { acc0[t][r] = 0.f; acc1[t][r] = 0.f; }
  for (int kk = 0; kk < HID + MLPD; kk += 32) {
    v16bf a0, a1;
    if (kk < HID) { a0 = load_a16(oR0, kk, lhalf); a1 = load_a16(oR1, kk, lhalf); }
    else          { a0 = load_a16(mR0, kk - HID, lhalf); a1 = load_a16(mR1, kk - HID, lhalf); }
    #pragma unroll
    for (int t = 0; t < 4; ++t) {
      const __bf16* bp = wt + (size_t)(n0 + t * 16 + lm) * (HID + MLPD);
      v16bf bfr = load_b16(bp, kk, lhalf);
      acc0[t] = wmma_bf16(a0, bfr, acc0[t]);
      acc1[t] = wmma_bf16(a1, bfr, acc1[t]);
    }
  }
  for (int t = 0; t < 4; ++t) {
    int col = n0 + t * 16 + lm;
    for (int r = 0; r < 8; ++r) {
      for (int g2 = 0; g2 < 2; ++g2) {
        int m = m0 + 16 * g2 + r + 8 * lhalf;
        int b = m >> 11, s = m & 2047;
        float hv = (g2 ? acc1[t][r] : acc0[t][r]) + bo[col];
        float g = emb[b * 3072 + 2048 + col];
        float res; float* dst;
        if (s < S_IMG) {
          size_t off = (size_t)(b * S_IMG + s) * HID + col;
          res = hid[off]; dst = out + off;
        } else {
          size_t off = (size_t)(b * S_TXT + (s - S_IMG)) * HID + col;
          res = enc[off]; dst = out + (size_t)Bn * S_IMG * HID + off;
        }
        *dst = g * hv + res;
      }
    }
  }
}

extern "C" void kernel_launch(void* const* d_in, const int* in_sizes, int n_in,
                              void* d_out, int out_size, void* d_ws, size_t ws_size,
                              hipStream_t stream) {
  (void)in_sizes; (void)n_in; (void)out_size; (void)ws_size;
  const float* hid  = (const float*)d_in[0];
  const float* enc  = (const float*)d_in[1];
  const float* temb = (const float*)d_in[2];
  const float* rc   = (const float*)d_in[3];
  const float* rsn  = (const float*)d_in[4];
  const float* cc   = (const float*)d_in[5];
  const float* csn  = (const float*)d_in[6];
  const float* wn   = (const float*)d_in[7];
  const float* bnv  = (const float*)d_in[8];
  const float* wq   = (const float*)d_in[9];
  const float* bq   = (const float*)d_in[10];
  const float* wk   = (const float*)d_in[11];
  const float* bk   = (const float*)d_in[12];
  const float* wv   = (const float*)d_in[13];
  const float* bv   = (const float*)d_in[14];
  const float* gq   = (const float*)d_in[15];
  const float* gk   = (const float*)d_in[16];
  const float* wm   = (const float*)d_in[17];
  const float* bm   = (const float*)d_in[18];
  const float* wo   = (const float*)d_in[19];
  const float* bo   = (const float*)d_in[20];

  char* ws = (char*)d_ws;
  float*  emb    = (float*)(ws + OFF_EMB);
  __bf16* nx     = (__bf16*)(ws + OFF_NX);
  __bf16* wqkv_t = (__bf16*)(ws + OFF_WQKV);
  __bf16* wmlp_t = (__bf16*)(ws + OFF_WMLP);
  __bf16* wout_t = (__bf16*)(ws + OFF_WOUT);
  float*  qkvf   = (float*)(ws + OFF_QKV);
  __bf16* qb     = (__bf16*)(ws + OFF_Q);
  __bf16* kb     = (__bf16*)(ws + OFF_K);
  __bf16* vtb    = (__bf16*)(ws + OFF_VT);
  __bf16* obf    = (__bf16*)(ws + OFF_O);
  __bf16* mlpb   = (__bf16*)(ws + OFF_MLPA);

  // weight conversion/transposition (fp32 [K,N] -> bf16 [N,K])
  k_transpose_bf16<<<(HID * HID) / 256, 256, 0, stream>>>(wq, wqkv_t,                 HID, HID);
  k_transpose_bf16<<<(HID * HID) / 256, 256, 0, stream>>>(wk, wqkv_t + HID * HID,     HID, HID);
  k_transpose_bf16<<<(HID * HID) / 256, 256, 0, stream>>>(wv, wqkv_t + 2 * HID * HID, HID, HID);
  k_transpose_bf16<<<(HID * MLPD) / 256, 256, 0, stream>>>(wm, wmlp_t, HID, MLPD);
  k_transpose_bf16<<<((HID + MLPD) * HID) / 256, 256, 0, stream>>>(wo, wout_t, HID + MLPD, HID);

  k_ada<<<(Bn * 3 * HID) / 256, 256, 0, stream>>>(temb, wn, bnv, emb);
  k_ln<<<TOK, 256, 0, stream>>>(hid, enc, emb, nx);

  k_gemm_qkv<<<(TOK / 32) * (3 * HID / 64) / 8, 256, 0, stream>>>(nx, wqkv_t, bq, bk, bv, qkvf);
  k_gemm_mlp<<<(TOK / 32) * (MLPD / 64) / 8, 256, 0, stream>>>(nx, wmlp_t, bm, mlpb);

  k_rmsrope<<<(Bn * Sq * Hh) / 8, 256, 0, stream>>>(qkvf, gq, gk, rc, rsn, cc, csn, qb, kb, vtb);

  k_attn<<<(Bn * Hh * (Sq / 16)) / 8, 256, 0, stream>>>(qb, kb, vtb, obf);

  k_gemm_out<<<(TOK / 32) * (HID / 64) / 8, 256, 0, stream>>>(obf, mlpb, wout_t, bo, emb,
                                                              hid, enc, (float*)d_out);
}
